// PointNet2feat_9758165696591
// MI455X (gfx1250) — matrix operand order repaired and checked by reading, not compiled, not run on hardware
//
#include <hip/hip_runtime.h>
#include <hip/hip_bf16.h>
#include <math.h>

typedef __attribute__((ext_vector_type(16))) _Float16 v16h;
typedef __attribute__((ext_vector_type(8)))  _Float16 v8h;
typedef __attribute__((ext_vector_type(8)))  float    v8f;

#define B_SZ 32
#define NPTS 2048
#define NSAMP 32

// ---------------- transpose (B,3,N) -> (B,N,3) ----------------
__global__ void transpose_kernel(const float* __restrict__ in, float* __restrict__ out, int N) {
  int t = blockIdx.x * blockDim.x + threadIdx.x;
  int total = B_SZ * N * 3;
  if (t >= total) return;
  int c = t % 3; int i = (t / 3) % N; int b = t / (3 * N);
  out[t] = in[(b * 3 + c) * N + i];
}

// ---------------- farthest point sampling: one block per batch -------------
// 1024 threads = 32 waves; per serial FPS step: <=2 pts/thread, wave32
// shfl-butterfly argmax, one 32-entry LDS round, 2 barriers total.
__global__ __launch_bounds__(1024) void fps_kernel(const float* __restrict__ xyz,
                                                   int N, int S,
                                                   float* __restrict__ new_xyz) {
  __shared__ float sx[NPTS], sy[NPTS], sz[NPTS], mind[NPTS];
  __shared__ float wv[32];
  __shared__ int   wi[32];
  __shared__ int   s_last;
  int b = blockIdx.x, tid = threadIdx.x;
  int lane = tid & 31, wid = tid >> 5;
  for (int i = tid; i < N; i += 1024) {
    sx[i] = xyz[(b * N + i) * 3 + 0];
    sy[i] = xyz[(b * N + i) * 3 + 1];
    sz[i] = xyz[(b * N + i) * 3 + 2];
    mind[i] = 1e10f;
  }
  __syncthreads();
  int last = 0;
  if (tid == 0) {
    new_xyz[(b * S + 0) * 3 + 0] = sx[0];
    new_xyz[(b * S + 0) * 3 + 1] = sy[0];
    new_xyz[(b * S + 0) * 3 + 2] = sz[0];
  }
  for (int it = 1; it < S; ++it) {
    float lx = sx[last], ly = sy[last], lz = sz[last];
    float bv = -1.0f; int bi = 0;
    for (int i = tid; i < N; i += 1024) {
      float dx = sx[i] - lx, dy = sy[i] - ly, dz = sz[i] - lz;
      float d = fminf(mind[i], dx * dx + dy * dy + dz * dz);
      mind[i] = d;
      if (d > bv) { bv = d; bi = i; }
    }
    // wave32 butterfly argmax (tie -> lower index, matching jnp.argmax)
#pragma unroll
    for (int m = 16; m > 0; m >>= 1) {
      float ov = __shfl_xor(bv, m, 32);
      int   oi = __shfl_xor(bi, m, 32);
      if (ov > bv || (ov == bv && oi < bi)) { bv = ov; bi = oi; }
    }
    if (lane == 0) { wv[wid] = bv; wi[wid] = bi; }
    __syncthreads();
    if (wid == 0) {
      float v2 = wv[lane]; int i2 = wi[lane];
#pragma unroll
      for (int m = 16; m > 0; m >>= 1) {
        float ov = __shfl_xor(v2, m, 32);
        int   oi = __shfl_xor(i2, m, 32);
        if (ov > v2 || (ov == v2 && oi < i2)) { v2 = ov; i2 = oi; }
      }
      if (lane == 0) {
        s_last = i2;
        new_xyz[(b * S + it) * 3 + 0] = sx[i2];
        new_xyz[(b * S + it) * 3 + 1] = sy[i2];
        new_xyz[(b * S + it) * 3 + 2] = sz[i2];
      }
    }
    __syncthreads();
    last = s_last;
  }
}

// ---------------- ball query: first NSAMP in-radius indices, pad w/ first ---
__global__ void ballq_kernel(const float* __restrict__ xyz,
                             const float* __restrict__ new_xyz,
                             int N, int S, float r2, int* __restrict__ nidx) {
  int q = blockIdx.x * blockDim.x + threadIdx.x;
  if (q >= B_SZ * S) return;
  int b = q / S;
  float cx = new_xyz[q * 3 + 0], cy = new_xyz[q * 3 + 1], cz = new_xyz[q * 3 + 2];
  int cnt = 0, first = 0;
  const float* base = xyz + (size_t)b * N * 3;
  for (int i = 0; i < N; ++i) {
    float dx = base[i * 3 + 0] - cx;
    float dy = base[i * 3 + 1] - cy;
    float dz = base[i * 3 + 2] - cz;
    float d2 = dx * dx + dy * dy + dz * dz;
    if (d2 < r2) {
      if (cnt == 0) first = i;
      nidx[q * NSAMP + cnt] = i;
      ++cnt;
      if (cnt >= NSAMP) break;
    }
  }
  for (int k = cnt; k < NSAMP; ++k) nidx[q * NSAMP + k] = first;
}

// ---------------- weight pack: (Cout,Cin) f32 -> WMMA-B fragment f16 -------
// block (kt,nt): 32 lanes x 16 halves contiguous; lane<16 -> K ktile*32+0..15,
// lane>=16 -> K ktile*32+16..31; column = ntile*16 + (lane&15). Zero-padded.
__global__ void pack_w_kernel(const float* __restrict__ w, const float* __restrict__ bias,
                              int Cin, int Cout, int Kpad, int Npad,
                              _Float16* __restrict__ ow, float* __restrict__ ob) {
  int t = blockIdx.x * blockDim.x + threadIdx.x;
  if (t < Npad) ob[t] = (t < Cout) ? bias[t] : 0.0f;
  int Ntiles = Npad >> 4;
  int total = Kpad * Npad;
  if (t >= total) return;
  int e    = t & 15;
  int lane = (t >> 4) & 31;
  int blk  = t >> 9;
  int nt = blk % Ntiles;
  int kt = blk / Ntiles;
  int k = kt * 32 + ((lane & 16) ? 16 : 0) + e;
  int n = nt * 16 + (lane & 15);
  float v = (k < Cin && n < Cout) ? w[n * Cin + k] : 0.0f;
  ow[t] = (_Float16)v;
}

// ---------------- one MLP layer: Y(32xNN) = relu(X(32xK) * W + b) ----------
template <int K, int NN>
__device__ __forceinline__ void mlp_layer(const _Float16* X, _Float16* Y,
                                          const _Float16* __restrict__ W,
                                          const float* __restrict__ bias,
                                          int wave, int lane) {
  const int mrow = wave * 16 + (lane & 15);
  const int koff = (lane & 16) ? 16 : 0;  // A-frag K-half (contiguous, matches packed B)
  const _Float16* xrow = X + mrow * K;
#pragma unroll 1
  for (int nt = 0; nt < NN / 16; ++nt) {
    v8f acc = {};
#pragma unroll
    for (int kt = 0; kt < K / 32; ++kt) {
      const _Float16* ap = xrow + kt * 32 + koff;
      v8h alo = *(const v8h*)(ap);        // K = base+0..7   (or 16..23)
      v8h ahi = *(const v8h*)(ap + 8);    // K = base+8..15  (or 24..31)
      v16h a = __builtin_shufflevector(alo, ahi, 0, 1, 2, 3, 4, 5, 6, 7,
                                       8, 9, 10, 11, 12, 13, 14, 15);
      v16h bm = *(const v16h*)(W + (size_t)((kt * (NN / 16) + nt) * 32 + lane) * 16);
      acc = __builtin_amdgcn_wmma_f32_16x16x32_f16(false, a, false, bm,
                                                   (short)0, acc, false, false);
    }
    int ncol = nt * 16 + (lane & 15);
    float bb = bias[ncol];
    int rbase = wave * 16 + ((lane & 16) ? 8 : 0);
#pragma unroll
    for (int r = 0; r < 8; ++r) {
      float v = acc[r] + bb;
      v = v > 0.0f ? v : 0.0f;            // ReLU
      Y[(rbase + r) * NN + ncol] = (_Float16)v;
    }
  }
}

// ------------- group + 3-layer MLP + neighbor-max; 1 workgroup per (b,s) ---
template <int CF, int K0, int N0, int K1, int N1, int K2, int N2, int COUT>
__global__ __launch_bounds__(64) void group_mlp_kernel(
    const float* __restrict__ xyz, const float* __restrict__ new_xyz,
    const int* __restrict__ nidx, const float* __restrict__ fin,
    int N, int S,
    const _Float16* __restrict__ w0, const float* __restrict__ b0,
    const _Float16* __restrict__ w1, const float* __restrict__ b1,
    const _Float16* __restrict__ w2, const float* __restrict__ b2,
    float* __restrict__ fout) {
  __shared__ __align__(32) _Float16 bufA[32 * 256];
  __shared__ __align__(32) _Float16 bufB[32 * 256];
  int q = blockIdx.x;
  int b = q / S;
  int tid = threadIdx.x, wave = tid >> 5, lane = tid & 31;
  float cx = new_xyz[q * 3 + 0], cy = new_xyz[q * 3 + 1], cz = new_xyz[q * 3 + 2];
  // build centered+concat input rows (32 x K0), zero-padded, as f16 in LDS
  for (int t = tid; t < 32 * K0; t += 64) {
    int r = t / K0, c = t - r * K0;
    int pi = nidx[q * NSAMP + r];
    float v = 0.0f;
    if (c < 3) {
      float p = xyz[((size_t)b * N + pi) * 3 + c];
      v = p - (c == 0 ? cx : (c == 1 ? cy : cz));
    } else if (c < 3 + CF) {
      v = fin[((size_t)b * N + pi) * CF + (c - 3)];
    }
    bufA[t] = (_Float16)v;
  }
  __syncthreads();
  mlp_layer<K0, N0>(bufA, bufB, w0, b0, wave, lane);
  __syncthreads();
  mlp_layer<K1, N1>(bufB, bufA, w1, b1, wave, lane);
  __syncthreads();
  mlp_layer<K2, N2>(bufA, bufB, w2, b2, wave, lane);
  __syncthreads();
  // max over the 32 neighbors
  for (int c = tid; c < COUT; c += 64) {
    float m = -1e30f;
#pragma unroll
    for (int r = 0; r < 32; ++r) m = fmaxf(m, (float)bufB[r * N2 + c]);
    fout[(size_t)q * COUT + c] = m;
  }
}

// ------------- post conv (170 out) + tanh + global max over S --------------
__global__ __launch_bounds__(256) void post_kernel(const float* __restrict__ feats,
                                                   int S, int C,
                                                   const float* __restrict__ pw,
                                                   const float* __restrict__ pb,
                                                   float* __restrict__ out, int ooff) {
  __shared__ float row[256];
  int b = blockIdx.x, tid = threadIdx.x;
  float best = -2.0f;
  for (int s = 0; s < S; ++s) {
    for (int c = tid; c < C; c += 256) row[c] = feats[((size_t)b * S + s) * C + c];
    __syncthreads();
    if (tid < 170) {
      float acc = pb[tid];
      const float* wrow = pw + (size_t)tid * C;
      for (int c = 0; c < C; ++c) acc = fmaf(row[c], wrow[c], acc);
      best = fmaxf(best, tanhf(acc));
    }
    __syncthreads();
  }
  if (tid < 170) out[(size_t)b * 510 + ooff + tid] = best;
}

extern "C" void kernel_launch(void* const* d_in, const int* in_sizes, int n_in,
                              void* d_out, int out_size, void* d_ws, size_t ws_size,
                              hipStream_t stream) {
  (void)in_sizes; (void)n_in; (void)out_size; (void)ws_size;
  const float* pc = (const float*)d_in[0];
  const float *w[3][3], *bs[3][3], *pw[3], *pb[3];
  int idx = 1;
  for (int l = 0; l < 3; ++l) {
    for (int j = 0; j < 3; ++j) { w[l][j] = (const float*)d_in[idx++]; bs[l][j] = (const float*)d_in[idx++]; }
    pw[l] = (const float*)d_in[idx++];
    pb[l] = (const float*)d_in[idx++];
  }
  float* out = (float*)d_out;

  // workspace carve-out
  char* ws = (char*)d_ws;
  size_t off = 0;
  auto alloc = [&](size_t bytes) -> char* {
    char* p = ws + off;
    off += (bytes + 255) & ~(size_t)255;
    return p;
  };
  float* xyz0 = (float*)alloc((size_t)B_SZ * 2048 * 3 * 4);
  float* nx0  = (float*)alloc((size_t)B_SZ * 512 * 3 * 4);
  float* nx1  = (float*)alloc((size_t)B_SZ * 128 * 3 * 4);
  float* nx2  = (float*)alloc((size_t)B_SZ * 32 * 3 * 4);
  int* ni0 = (int*)alloc((size_t)B_SZ * 512 * NSAMP * 4);
  int* ni1 = (int*)alloc((size_t)B_SZ * 128 * NSAMP * 4);
  int* ni2 = (int*)alloc((size_t)B_SZ * 32 * NSAMP * 4);
  float* f0 = (float*)alloc((size_t)B_SZ * 512 * 96 * 4);
  float* f1 = (float*)alloc((size_t)B_SZ * 128 * 192 * 4);
  float* f2 = (float*)alloc((size_t)B_SZ * 32 * 256 * 4);

  static const int KP[3][3]  = {{32, 32, 64}, {128, 64, 96}, {224, 96, 192}};
  static const int NP[3][3]  = {{32, 64, 96}, {64, 96, 192}, {96, 192, 256}};
  static const int CI[3][3]  = {{3, 24, 48}, {99, 48, 96}, {195, 96, 192}};
  static const int CO[3][3]  = {{24, 48, 96}, {48, 96, 192}, {96, 192, 256}};
  _Float16* wp[3][3];
  float* bp[3][3];
  for (int l = 0; l < 3; ++l)
    for (int j = 0; j < 3; ++j) {
      wp[l][j] = (_Float16*)alloc((size_t)KP[l][j] * NP[l][j] * 2);
      bp[l][j] = (float*)alloc((size_t)NP[l][j] * 4);
    }

  // transpose input to (B,N,3)
  {
    int total = B_SZ * 2048 * 3;
    transpose_kernel<<<(total + 255) / 256, 256, 0, stream>>>(pc, xyz0, 2048);
  }
  // pack all 9 weight matrices + padded biases
  for (int l = 0; l < 3; ++l)
    for (int j = 0; j < 3; ++j) {
      int total = KP[l][j] * NP[l][j];
      pack_w_kernel<<<(total + 255) / 256, 256, 0, stream>>>(
          w[l][j], bs[l][j], CI[l][j], CO[l][j], KP[l][j], NP[l][j], wp[l][j], bp[l][j]);
    }

  // ---------------- level 0 ----------------
  fps_kernel<<<B_SZ, 1024, 0, stream>>>(xyz0, 2048, 512, nx0);
  ballq_kernel<<<(B_SZ * 512 + 255) / 256, 256, 0, stream>>>(xyz0, nx0, 2048, 512, 0.04f, ni0);
  group_mlp_kernel<0, 32, 32, 32, 64, 64, 96, 96><<<B_SZ * 512, 64, 0, stream>>>(
      xyz0, nx0, ni0, xyz0, 2048, 512,
      wp[0][0], bp[0][0], wp[0][1], bp[0][1], wp[0][2], bp[0][2], f0);
  post_kernel<<<B_SZ, 256, 0, stream>>>(f0, 512, 96, pw[0], pb[0], out, 0);

  // ---------------- level 1 ----------------
  fps_kernel<<<B_SZ, 1024, 0, stream>>>(nx0, 512, 128, nx1);
  ballq_kernel<<<(B_SZ * 128 + 255) / 256, 256, 0, stream>>>(nx0, nx1, 512, 128, 0.16f, ni1);
  group_mlp_kernel<96, 128, 64, 64, 96, 96, 192, 192><<<B_SZ * 128, 64, 0, stream>>>(
      nx0, nx1, ni1, f0, 512, 128,
      wp[1][0], bp[1][0], wp[1][1], bp[1][1], wp[1][2], bp[1][2], f1);
  post_kernel<<<B_SZ, 256, 0, stream>>>(f1, 128, 192, pw[1], pb[1], out, 170);

  // ---------------- level 2 ----------------
  fps_kernel<<<B_SZ, 1024, 0, stream>>>(nx1, 128, 32, nx2);
  ballq_kernel<<<(B_SZ * 32 + 255) / 256, 256, 0, stream>>>(nx1, nx2, 128, 32, 0.64f, ni2);
  group_mlp_kernel<192, 224, 96, 96, 192, 192, 256, 256><<<B_SZ * 32, 64, 0, stream>>>(
      nx1, nx2, ni2, f1, 128, 32,
      wp[2][0], bp[2][0], wp[2][1], bp[2][1], wp[2][2], bp[2][2], f2);
  post_kernel<<<B_SZ, 256, 0, stream>>>(f2, 32, 256, pw[2], pb[2], out, 340);
}
